// MoonViT3dEncoder_18545668784287
// MI455X (gfx1250) — compile-verified, matrix operand order checked by reading, not run on hardware
//
#include <hip/hip_runtime.h>
#include <hip/hip_bf16.h>

// ---------------------------------------------------------------------------
// MoonViT 3D encoder for gfx1250 (MI455X).
// All matmuls run on v_wmma_f32_16x16x32_bf16 (bf16 inputs, fp32 accumulate).
// GEMM: 8 waves/block, 256x64 block tile, double-buffered LDS B staging,
// one barrier per k-step, 8 WMMAs per wave per k-step.
// ---------------------------------------------------------------------------

typedef __bf16 bf16;
typedef __attribute__((ext_vector_type(16))) __bf16 bf16x16;
typedef __attribute__((ext_vector_type(8)))  float  fx8;

#define WMMA_BF16(a, b, c) \
    __builtin_amdgcn_wmma_f32_16x16x32_bf16(false, (a), false, (b), (short)0, (c), false, false)

// ------------------------------ fp32 -> bf16 -------------------------------
__global__ void moonvit_cvt_bf16(const float* __restrict__ in, bf16* __restrict__ out, size_t n) {
    size_t i = (size_t)blockIdx.x * blockDim.x + threadIdx.x;
    if (i < n) out[i] = (bf16)in[i];
}

// ------------------------------- LayerNorm ---------------------------------
// One block (256 threads, 8 waves) per row of H=1024. Optional fp32 and bf16 outputs.
__global__ __launch_bounds__(256)
void moonvit_layernorm(const float* __restrict__ x,
                       const float* __restrict__ g,
                       const float* __restrict__ b,
                       float* __restrict__ outF,
                       bf16* __restrict__ outB,
                       int Hdim) {
    const int row  = blockIdx.x;
    const int tid  = threadIdx.x;
    const int lane = tid & 31;
    const int wid  = tid >> 5;

    const float4* xr = reinterpret_cast<const float4*>(x + (size_t)row * Hdim);
    float4 v = xr[tid];
    float s  = v.x + v.y + v.z + v.w;
    float ss = v.x * v.x + v.y * v.y + v.z * v.z + v.w * v.w;

    #pragma unroll
    for (int m = 1; m < 32; m <<= 1) {
        s  += __shfl_xor(s,  m, 32);
        ss += __shfl_xor(ss, m, 32);
    }
    __shared__ float ps[8], pss[8];
    if (lane == 0) { ps[wid] = s; pss[wid] = ss; }
    __syncthreads();
    float ts = 0.f, tss = 0.f;
    #pragma unroll
    for (int w = 0; w < 8; ++w) { ts += ps[w]; tss += pss[w]; }

    const float invH = 1.0f / (float)Hdim;
    const float mu   = ts * invH;
    const float var  = tss * invH - mu * mu;
    const float rstd = rsqrtf(var + 1e-5f);

    const float4 gv = reinterpret_cast<const float4*>(g)[tid];
    const float4 bv = reinterpret_cast<const float4*>(b)[tid];
    float o0 = (v.x - mu) * rstd * gv.x + bv.x;
    float o1 = (v.y - mu) * rstd * gv.y + bv.y;
    float o2 = (v.z - mu) * rstd * gv.z + bv.z;
    float o3 = (v.w - mu) * rstd * gv.w + bv.w;

    const size_t base = (size_t)row * Hdim + tid * 4;
    if (outF) {
        float4 ov = make_float4(o0, o1, o2, o3);
        reinterpret_cast<float4*>(outF + (size_t)row * Hdim)[tid] = ov;
    }
    if (outB) {
        outB[base + 0] = (bf16)o0;
        outB[base + 1] = (bf16)o1;
        outB[base + 2] = (bf16)o2;
        outB[base + 3] = (bf16)o3;
    }
}

// ----------------------------- WMMA GEMM -----------------------------------
// C[M,N] = A[M,K](bf16, row-major) @ W[N,K]^T (bf16, row-major) [+ epilogue]
// Block: 256 threads = 8 waves. Block tile = 256 rows x 64 cols
// (each wave: 32 rows x 64 cols = two A fragments x four B tiles = 8 WMMA/k-step).
// B tile (64 cols x 32 k, 4 KB) double-buffered in LDS: one barrier per k-step.
// modes: 0 = plain f32 out, 1 = +bias, GELU -> bf16 out,
//        2 = +residual -> f32 out, 3 = +bias +residual -> f32 out
__global__ __launch_bounds__(256)
void moonvit_gemm_bf16(const bf16* __restrict__ A, int lda,
                       const bf16* __restrict__ W, int ldw,
                       int K,
                       float* __restrict__ outF, bf16* __restrict__ outB,
                       const float* __restrict__ bias,
                       const float* __restrict__ res,
                       int ldc, int mode) {
    __shared__ __align__(64) bf16 Bs[2][64 * 32];

    const int tid  = threadIdx.x;
    const int wid  = tid >> 5;
    const int lane = tid & 31;
    const int l16  = lane & 15;
    const int kh   = lane >> 4;

    const int rowbase = blockIdx.x * 256 + wid * 32;
    const int colbase = blockIdx.y * 64;

    fx8 acc[2][4] = {};

    const bf16* Arow0 = A + (size_t)(rowbase + l16) * lda + kh * 16;
    const bf16* Arow1 = Arow0 + (size_t)16 * lda;

    // cooperative B staging: thread t copies 16 B of W[colbase + t/4][k + (t%4)*8 ..]
    const int sc = tid >> 2;
    const int sp = tid & 3;
    const bf16* Wsrc = W + (size_t)(colbase + sc) * ldw + sp * 8;
    uint4* BsDst0 = reinterpret_cast<uint4*>(&Bs[0][sc * 32 + sp * 8]);
    uint4* BsDst1 = reinterpret_cast<uint4*>(&Bs[1][sc * 32 + sp * 8]);

    // preload first B tile
    *BsDst0 = *reinterpret_cast<const uint4*>(Wsrc);
    __syncthreads();

    for (int k = 0; k < K; k += 32) {
        const int cur = (k >> 5) & 1;
        // stage next B tile into the other buffer while computing this one
        if (k + 32 < K) {
            uint4* dst = cur ? BsDst0 : BsDst1;
            *dst = *reinterpret_cast<const uint4*>(Wsrc + k + 32);
        }
        __builtin_prefetch(Arow0 + k + 32, 0, 1);   // global_prefetch_b8

        const bf16x16 a0 = *reinterpret_cast<const bf16x16*>(Arow0 + k);
        const bf16x16 a1 = *reinterpret_cast<const bf16x16*>(Arow1 + k);
        #pragma unroll
        for (int j = 0; j < 4; ++j) {
            const bf16x16 bfr =
                *reinterpret_cast<const bf16x16*>(&Bs[cur][(j * 16 + l16) * 32 + kh * 16]);
            acc[0][j] = WMMA_BF16(a0, bfr, acc[0][j]);
            acc[1][j] = WMMA_BF16(a1, bfr, acc[1][j]);
        }
        __syncthreads();   // separates this iter's reads of Bs[cur] from next overwrite
    }

    #pragma unroll
    for (int rt = 0; rt < 2; ++rt) {
        #pragma unroll
        for (int j = 0; j < 4; ++j) {
            #pragma unroll
            for (int r = 0; r < 8; ++r) {
                const int row = rowbase + rt * 16 + r + 8 * kh;
                const int col = colbase + j * 16 + l16;
                float v = acc[rt][j][r];
                if (mode == 1 || mode == 3) v += bias[col];
                if (mode == 1) {
                    v = 0.5f * v * (1.0f + erff(v * 0.70710678118654752f));  // exact GELU
                    outB[(size_t)row * ldc + col] = (bf16)v;
                } else if (mode == 2 || mode == 3) {
                    v += res[(size_t)row * ldc + col];
                    outF[(size_t)row * ldc + col] = v;
                } else {
                    outF[(size_t)row * ldc + col] = v;
                }
            }
        }
    }
}

// ------------------------------- RoPE + repack -----------------------------
// qkv f32 [S][3][NH=16][HD=64] ->
//   Qh,Kh bf16 [NH][S][HD] (rope applied), Vt bf16 [NH][HD][S] (transposed).
// One thread per (s, head, pair p in 0..31).
__global__ void moonvit_rope(const float* __restrict__ qkv,
                             const int* __restrict__ grids,
                             bf16* __restrict__ Qh, bf16* __restrict__ Kh,
                             bf16* __restrict__ Vt, int S) {
    const int id    = blockIdx.x * blockDim.x + threadIdx.x;
    const int total = S * 16 * 32;
    if (id >= total) return;
    const int p    = id & 31;
    const int head = (id >> 5) & 15;
    const int s    = id >> 9;

    // 3-D grid position (2 segments, as in the reference)
    const int len0 = grids[0] * grids[1] * grids[2];
    int xi, yi;
    if (s < len0) {
        const int hw = grids[1] * grids[2];
        const int local = s % hw;
        yi = local / grids[2];
        xi = local % grids[2];
    } else {
        const int hw = grids[4] * grids[5];
        const int local = (s - len0) % hw;
        yi = local / grids[5];
        xi = local % grids[5];
    }

    const int   fi    = p >> 1;
    const float freq  = __powf(10000.0f, -(float)fi * (1.0f / 16.0f));
    const float coord = (p & 1) ? (float)yi : (float)xi;
    float sn, cs;
    __sincosf(coord * freq, &sn, &cs);

    const size_t base = (size_t)s * 3072 + head * 64 + p * 2;  // which=0 (q)
    const float qa = qkv[base],        qb = qkv[base + 1];
    const float ka = qkv[base + 1024], kb = qkv[base + 1025];
    const float va = qkv[base + 2048], vb = qkv[base + 2049];

    const size_t qo = ((size_t)head * S + s) * 64 + p * 2;
    Qh[qo]     = (bf16)(qa * cs - qb * sn);
    Qh[qo + 1] = (bf16)(qa * sn + qb * cs);
    Kh[qo]     = (bf16)(ka * cs - kb * sn);
    Kh[qo + 1] = (bf16)(ka * sn + kb * cs);

    const size_t vo = ((size_t)head * 64 + p * 2) * S + s;
    Vt[vo]     = (bf16)va;
    Vt[vo + S] = (bf16)vb;
}

// ------------------------------ Attention ----------------------------------
// One wave (32 threads) per (head, 16-query tile). Flash-style online softmax.
// Segment mask handled exactly by key-range restriction (segments 32-aligned).
// P repack (WMMA C layout -> A layout) double-buffered in LDS: one barrier/chunk.
__global__ __launch_bounds__(32)
void moonvit_attention(const bf16* __restrict__ Qh, const bf16* __restrict__ Kh,
                       const bf16* __restrict__ Vt, const int* __restrict__ grids,
                       bf16* __restrict__ O, int S, int Hdim) {
    __shared__ __align__(64) bf16 Pls[2][16 * 32];

    const int lane = threadIdx.x;
    const int l16  = lane & 15;
    const int kh   = lane >> 4;
    const int head = blockIdx.y;
    const int row0 = blockIdx.x * 16;

    const int len0 = grids[0] * grids[1] * grids[2];
    const int len1 = grids[3] * grids[4] * grids[5];
    int kbeg, kend;
    if (row0 < len0) { kbeg = 0;    kend = len0;        }
    else             { kbeg = len0; kend = len0 + len1; }

    const size_t qoff = ((size_t)head * S + row0 + l16) * 64 + kh * 16;
    const bf16x16 qa0 = *reinterpret_cast<const bf16x16*>(Qh + qoff);        // hd 0..31
    const bf16x16 qa1 = *reinterpret_cast<const bf16x16*>(Qh + qoff + 32);   // hd 32..63

    fx8 oacc[4] = {};
    float m[8], lsum[8];
    #pragma unroll
    for (int r = 0; r < 8; ++r) { m[r] = -3.0e38f; lsum[r] = 0.0f; }

    const float scale = 0.125f;  // 1/sqrt(64)
    int buf = 0;

    for (int kc = kbeg; kc < kend; kc += 32, buf ^= 1) {
        // ---- scores: S_tile[16q x 32k] = Q(16x64) @ K^T, two 16x16 tiles ----
        fx8 s0 = {}, s1 = {};
        {
            size_t koff = ((size_t)head * S + kc + l16) * 64 + kh * 16;
            bf16x16 b0 = *reinterpret_cast<const bf16x16*>(Kh + koff);
            bf16x16 b1 = *reinterpret_cast<const bf16x16*>(Kh + koff + 32);
            s0 = WMMA_BF16(qa0, b0, s0);
            s0 = WMMA_BF16(qa1, b1, s0);
            koff = ((size_t)head * S + kc + 16 + l16) * 64 + kh * 16;
            b0 = *reinterpret_cast<const bf16x16*>(Kh + koff);
            b1 = *reinterpret_cast<const bf16x16*>(Kh + koff + 32);
            s1 = WMMA_BF16(qa0, b0, s1);
            s1 = WMMA_BF16(qa1, b1, s1);
        }

        // ---- online softmax (C layout: row = r + 8*kh, col = l16) ----
        float p0[8], p1[8], ratio[8];
        #pragma unroll
        for (int r = 0; r < 8; ++r) {
            const float a = s0[r] * scale;
            const float b = s1[r] * scale;
            float mx = fmaxf(a, b);
            mx = fmaxf(mx, __shfl_xor(mx, 1, 32));
            mx = fmaxf(mx, __shfl_xor(mx, 2, 32));
            mx = fmaxf(mx, __shfl_xor(mx, 4, 32));
            mx = fmaxf(mx, __shfl_xor(mx, 8, 32));
            const float mn = fmaxf(m[r], mx);
            ratio[r] = __expf(m[r] - mn);
            m[r] = mn;
            p0[r] = __expf(a - mn);
            p1[r] = __expf(b - mn);
            float rs = p0[r] + p1[r];
            rs += __shfl_xor(rs, 1, 32);
            rs += __shfl_xor(rs, 2, 32);
            rs += __shfl_xor(rs, 4, 32);
            rs += __shfl_xor(rs, 8, 32);
            lsum[r] = lsum[r] * ratio[r] + rs;
        }
        #pragma unroll
        for (int j = 0; j < 4; ++j)
            #pragma unroll
            for (int r = 0; r < 8; ++r)
                oacc[j][r] *= ratio[r];

        // ---- repack P: C layout -> A layout via double-buffered LDS bounce ----
        #pragma unroll
        for (int r = 0; r < 8; ++r) {
            const int prow = r + 8 * kh;
            Pls[buf][prow * 32 + l16]      = (bf16)p0[r];
            Pls[buf][prow * 32 + 16 + l16] = (bf16)p1[r];
        }
        __syncthreads();   // buf is not overwritten until two iterations later
        const bf16x16 pa = *reinterpret_cast<const bf16x16*>(&Pls[buf][l16 * 32 + kh * 16]);

        // ---- O += P(16x32) @ V(32x64); V transposed so lanes load contiguous keys ----
        #pragma unroll
        for (int j = 0; j < 4; ++j) {
            const bf16x16 bv = *reinterpret_cast<const bf16x16*>(
                Vt + ((size_t)head * 64 + j * 16 + l16) * S + kc + kh * 16);
            oacc[j] = WMMA_BF16(pa, bv, oacc[j]);
        }
    }

    float inv[8];
    #pragma unroll
    for (int r = 0; r < 8; ++r) inv[r] = 1.0f / lsum[r];
    #pragma unroll
    for (int j = 0; j < 4; ++j) {
        #pragma unroll
        for (int r = 0; r < 8; ++r) {
            const int row = row0 + r + 8 * kh;
            const int col = head * 64 + j * 16 + l16;
            O[(size_t)row * Hdim + col] = (bf16)(oacc[j][r] * inv[r]);
        }
    }
}

// ------------------------------- host side ---------------------------------
extern "C" void kernel_launch(void* const* d_in, const int* in_sizes, int n_in,
                              void* d_out, int out_size, void* d_ws, size_t ws_size,
                              hipStream_t stream) {
    const int H = 1024, NH = 16, HD = 64, Mm = 4096, Lh = 2;
    const int threeH = 3 * H;
    const int S = in_sizes[0] / H;  // 3072

    const float* hidden = (const float*)d_in[0];
    const int*   grids  = (const int*)d_in[1];
    const float* ln0_g  = (const float*)d_in[2];
    const float* ln0_b  = (const float*)d_in[3];
    const float* wqkv   = (const float*)d_in[4];
    const float* wo     = (const float*)d_in[5];
    const float* ln1_g  = (const float*)d_in[6];
    const float* ln1_b  = (const float*)d_in[7];
    const float* fc0_w  = (const float*)d_in[8];
    const float* fc0_b  = (const float*)d_in[9];
    const float* fc1_w  = (const float*)d_in[10];
    const float* fc1_b  = (const float*)d_in[11];
    const float* fln_g  = (const float*)d_in[12];
    const float* fln_b  = (const float*)d_in[13];
    float* out = (float*)d_out;

    // workspace layout
    char*  ws  = (char*)d_ws;
    size_t off = 0;
    auto take = [&](size_t bytes) -> char* {
        char* p = ws + off;
        off = (off + bytes + 255) & ~(size_t)255;
        return p;
    };
    bf16* wqkv_bf = (bf16*)take((size_t)Lh * threeH * H * 2);
    bf16* wo_bf   = (bf16*)take((size_t)Lh * H * H * 2);
    bf16* fc0_bf  = (bf16*)take((size_t)Lh * Mm * H * 2);
    bf16* fc1_bf  = (bf16*)take((size_t)Lh * H * Mm * 2);
    float* x      = (float*)take((size_t)S * H * 4);
    bf16* hbf     = (bf16*)take((size_t)S * H * 2);
    float* qkv    = (float*)take((size_t)S * threeH * 4);
    bf16*  mlp    = (bf16*)qkv;                 // lifetime-disjoint reuse
    bf16* qh      = (bf16*)take((size_t)S * H * 2);
    bf16* kh      = (bf16*)take((size_t)S * H * 2);
    bf16* vt      = (bf16*)take((size_t)S * H * 2);
    bf16* obf     = (bf16*)take((size_t)S * H * 2);
    (void)ws_size; (void)n_in; (void)out_size;

    // --- convert all weights to bf16 (both layers per launch) ---
    auto cvt = [&](const float* src, bf16* dst, size_t n) {
        moonvit_cvt_bf16<<<dim3((unsigned)((n + 255) / 256)), dim3(256), 0, stream>>>(src, dst, n);
    };
    cvt(wqkv,  wqkv_bf, (size_t)Lh * threeH * H);
    cvt(wo,    wo_bf,   (size_t)Lh * H * H);
    cvt(fc0_w, fc0_bf,  (size_t)Lh * Mm * H);
    cvt(fc1_w, fc1_bf,  (size_t)Lh * H * Mm);

    // x = hidden_states
    hipMemcpyAsync(x, hidden, (size_t)S * H * 4, hipMemcpyDeviceToDevice, stream);

    const dim3 blk256(256), blk32(32);
    for (int i = 0; i < Lh; ++i) {
        // --- LN0 -> hbf (bf16) ---
        moonvit_layernorm<<<dim3(S), blk256, 0, stream>>>(
            x, ln0_g + (size_t)i * H, ln0_b + (size_t)i * H, nullptr, hbf, H);

        // --- qkv = hbf @ wqkv^T (f32 out) ---
        moonvit_gemm_bf16<<<dim3(S / 256, threeH / 64), blk256, 0, stream>>>(
            hbf, H, wqkv_bf + (size_t)i * threeH * H, H, H,
            qkv, nullptr, nullptr, nullptr, threeH, 0);

        // --- RoPE + repack into Qh/Kh [NH,S,HD], Vt [NH,HD,S] ---
        {
            const int total = S * NH * 32;
            moonvit_rope<<<dim3((total + 255) / 256), blk256, 0, stream>>>(
                qkv, grids, qh, kh, vt, S);
        }

        // --- attention -> obf (bf16 [S,H]) ---
        moonvit_attention<<<dim3(S / 16, NH), blk32, 0, stream>>>(
            qh, kh, vt, grids, obf, S, H);

        // --- x = x + obf @ wo^T ---
        moonvit_gemm_bf16<<<dim3(S / 256, H / 64), blk256, 0, stream>>>(
            obf, H, wo_bf + (size_t)i * H * H, H, H,
            x, nullptr, nullptr, x, H, 2);

        // --- LN1 -> hbf ---
        moonvit_layernorm<<<dim3(S), blk256, 0, stream>>>(
            x, ln1_g + (size_t)i * H, ln1_b + (size_t)i * H, nullptr, hbf, H);

        // --- mlp = gelu(hbf @ fc0^T + fc0_b) (bf16 out) ---
        moonvit_gemm_bf16<<<dim3(S / 256, Mm / 64), blk256, 0, stream>>>(
            hbf, H, fc0_bf + (size_t)i * Mm * H, H, H,
            nullptr, mlp, fc0_b + (size_t)i * Mm, nullptr, Mm, 1);

        // --- x = x + mlp @ fc1^T + fc1_b ---
        moonvit_gemm_bf16<<<dim3(S / 256, H / 64), blk256, 0, stream>>>(
            mlp, Mm, fc1_bf + (size_t)i * H * Mm, Mm, Mm,
            x, nullptr, fc1_b + (size_t)i * H, x, H, 3);
    }

    // --- final LayerNorm -> d_out (f32) ---
    moonvit_layernorm<<<dim3(S), blk256, 0, stream>>>(
        x, fln_g, fln_b, out, nullptr, H);
}
